// CoDA_57732950393267
// MI455X (gfx1250) — matrix operand order repaired and checked by reading, not compiled, not run on hardware
//
#include <hip/hip_runtime.h>
#include <hip/hip_bf16.h>
#include <math.h>

// ---------------------------------------------------------------------------
// CoDA attention for MI455X (gfx1250, wave32, WMMA + double-buffered TDM)
//   q,k,v = X @ W^T (bf16 WMMA, f32 acc)
//   E = q k^T (WMMA);  N = sum_d |q-k| (VALU from LDS, f32; K tiles via TDM)
//   coda = tanh(E)*sigmoid(N);  out = coda @ v (WMMA)
//   fc + bias + residual (WMMA) -> LayerNorm
// ---------------------------------------------------------------------------

typedef __bf16 bf16_t;
typedef __attribute__((ext_vector_type(16))) __bf16       v16bf;
typedef __attribute__((ext_vector_type(8)))  float        v8f;
typedef __attribute__((ext_vector_type(4)))  unsigned int u32x4;
typedef __attribute__((ext_vector_type(8)))  int          i32x8;
typedef __attribute__((ext_vector_type(4)))  int          i32x4;

#define DMODEL 1024
#define NHEADS 16
#define HDIM   64
#define BATCH  4
#define SEQ    1024

__device__ __forceinline__ bf16_t to_bf16(float f) {
  union { float f; unsigned int u; } v; v.f = f;
  unsigned int r = v.u + 0x7FFFu + ((v.u >> 16) & 1u);   // RNE
  union { unsigned short s; bf16_t b; } o; o.s = (unsigned short)(r >> 16);
  return o.b;
}

// A-frag (16x32 bf16, M x K): lane<16 -> M=lane, K chunks {k0..k0+7, k0+16..k0+23}
//                             lane>=16 -> M=lane-16, chunks shifted by +8 in K.
__device__ __forceinline__ v16bf load_frag_a(const bf16_t* base, int ld,
                                             int row0, int k0, int lane) {
  const int m    = lane & 15;
  const int koff = (lane < 16) ? 0 : 8;
  const bf16_t* p = base + (size_t)(row0 + m) * ld + k0 + koff;
  union { v16bf v; u32x4 q[2]; } u;
  u.q[0] = *(const u32x4*)(p);
  u.q[1] = *(const u32x4*)(p + 16);
  return u.v;
}

// B-frag (32x16 bf16, K x N) from a source stored N-major along K:
// lane<16 -> N=lane, K chunks {k0..k0+7, k0+16..k0+23}; lane>=16 -> N=lane-16, chunks shifted by +8.
__device__ __forceinline__ v16bf load_frag_b(const bf16_t* base, int ld,
                                             int col0, int k0, int lane) {
  const int n    = lane & 15;
  const int koff = (lane < 16) ? 0 : 8;
  const bf16_t* p = base + (size_t)(col0 + n) * ld + k0 + koff;
  union { v16bf v; u32x4 q[2]; } u;
  u.q[0] = *(const u32x4*)(p);
  u.q[1] = *(const u32x4*)(p + 16);
  return u.v;
}

#define WMMA_BF16(A, B, C) \
  __builtin_amdgcn_wmma_f32_16x16x32_bf16(false, (A), false, (B), (short)0, (C), false, false)

// Issue a TDM load of a 32x64 f32 tile (row stride DMODEL) into LDS at
// lds_off, padding 1 DWORD after every 64 DWORDs -> [32][65] layout.
__device__ __forceinline__ void tdm_load_k_tile(const float* gsrc,
                                                unsigned int lds_off) {
  const unsigned long long ga = (unsigned long long)(uintptr_t)gsrc;
  u32x4 g0;
  g0[0] = 1u;                                        // count=1, user mode
  g0[1] = lds_off;                                   // lds_addr
  g0[2] = (unsigned int)(ga & 0xFFFFFFFFu);          // global_addr[31:0]
  g0[3] = (unsigned int)((ga >> 32) & 0x01FFFFFFu)   // global_addr[56:32]
        | (2u << 30);                                // type=2 (image)
  i32x8 g1;
  g1[0] = (int)((2u << 16) | (1u << 20) | (5u << 22)); // 4B elems, pad_en,
                                                       // pad 1 DW per 64 DW
  g1[1] = (int)(64u << 16);                          // tensor_dim0 = 64
  g1[2] = (int)(32u << 16);                          // tensor_dim1 = 32
  g1[3] = (int)(64u << 16);                          // tile_dim0 = 64
  g1[4] = 32;                                        // tile_dim1 = 32
  g1[5] = DMODEL;                                    // tensor_dim0_stride
  g1[6] = 0; g1[7] = 0;
  const i32x4 z4 = {0, 0, 0, 0};
  const i32x8 z8 = {0, 0, 0, 0, 0, 0, 0, 0};
  __builtin_amdgcn_tensor_load_to_lds(g0, g1, z4, z4, z8, 0);
}

// ---------------------------------------------------------------------------
// fp32 -> bf16 conversion
// ---------------------------------------------------------------------------
__global__ void k_cvt_bf16(const float* __restrict__ src,
                           bf16_t* __restrict__ dst, int n) {
  int i = blockIdx.x * blockDim.x + threadIdx.x;
  if (i < n) dst[i] = to_bf16(src[i]);
}

// ---------------------------------------------------------------------------
// C[M,N] = A[M,K] @ B[N,K]^T  (bf16 in, f32 acc), optional bias/residual,
// optional f32 and/or bf16 outputs.
// Block = 8 waves, block tile 64x128; each wave a 32x32 register tile
// (2 A-frags x 2 B-frags -> 4 WMMA per k-step).
// ---------------------------------------------------------------------------
__global__ __launch_bounds__(256) void k_gemm_nt(
    const bf16_t* __restrict__ A, const bf16_t* __restrict__ Bm,
    int M, int N, int K,
    float* __restrict__ Cf, bf16_t* __restrict__ Cb,
    const float* __restrict__ bias, const float* __restrict__ resid) {
  const int lane = threadIdx.x & 31;
  const int w    = threadIdx.x >> 5;
  const int row0 = blockIdx.y * 64  + (w >> 2) * 32;
  const int col0 = blockIdx.x * 128 + (w & 3) * 32;

  v8f acc00 = {}, acc01 = {}, acc10 = {}, acc11 = {};
  for (int k0 = 0; k0 < K; k0 += 32) {
    if (k0 + 64 <= K) {   // prefetch next K tiles into the SE cache
      __builtin_prefetch(A  + (size_t)row0 * K + k0 + 32, 0, 3);
      __builtin_prefetch(Bm + (size_t)col0 * K + k0 + 32, 0, 3);
    }
    v16bf a0 = load_frag_a(A,  K, row0,      k0, lane);
    v16bf a1 = load_frag_a(A,  K, row0 + 16, k0, lane);
    v16bf b0 = load_frag_b(Bm, K, col0,      k0, lane);
    v16bf b1 = load_frag_b(Bm, K, col0 + 16, k0, lane);
    acc00 = WMMA_BF16(a0, b0, acc00);
    acc01 = WMMA_BF16(a0, b1, acc01);
    acc10 = WMMA_BF16(a1, b0, acc10);
    acc11 = WMMA_BF16(a1, b1, acc11);
  }

  const int mloc = (lane >> 4) << 3;     // C layout: M = r + 8*(lane/16)
  const int nl   = lane & 15;
  #pragma unroll
  for (int r = 0; r < 8; ++r) {
    #pragma unroll
    for (int t = 0; t < 4; ++t) {
      const int m = row0 + ((t >> 1) ? 16 : 0) + mloc + r;
      const int n = col0 + ((t & 1)  ? 16 : 0) + nl;
      float v = (t == 0) ? acc00[r] : (t == 1) ? acc01[r]
              : (t == 2) ? acc10[r] : acc11[r];
      if (bias)  v += bias[n];
      if (resid) v += resid[(size_t)m * N + n];
      if (Cf) Cf[(size_t)m * N + n] = v;
      if (Cb) Cb[(size_t)m * N + n] = to_bf16(v);
    }
  }
}

// ---------------------------------------------------------------------------
// Fused CoDA attention: per block = 128 query rows of one (b,h);
// 8 waves x one 16-row i-tile each; stream j in 32-row K/V tiles.
// K f32 tiles are double-buffered through the Tensor Data Mover:
// tile js+1 is in flight while tile js is consumed (s_wait_tensorcnt 1).
// ---------------------------------------------------------------------------
__global__ __launch_bounds__(256) void k_coda_attn(
    const float*  __restrict__ Qf, const float*  __restrict__ Kf,
    const bf16_t* __restrict__ Qb, const bf16_t* __restrict__ Kb,
    const bf16_t* __restrict__ Vb, bf16_t* __restrict__ Ob) {
  __shared__ float  q_lds[128][HDIM];          // 32 KB (f32 q rows for N)
  __shared__ float  k_lds[2][32][HDIM + 1];    // 16.6 KB (TDM double buffer)
  __shared__ bf16_t vT[HDIM][40];              // 5 KB  (v tile, d-major)
  __shared__ bf16_t coda_lds[8][16][32];       // 8 KB  (per-wave coda staging)

  const int lane = threadIdx.x & 31;
  const int w    = threadIdx.x >> 5;
  const int bh   = blockIdx.y;
  const int b    = bh / NHEADS;
  const int h    = bh % NHEADS;
  const int row0 = blockIdx.x * 128;
  const size_t base = (size_t)b * SEQ * DMODEL + (size_t)h * HDIM;

  // stage this block's 128 query rows (head slice, f32)
  for (int idx = threadIdx.x; idx < 128 * HDIM; idx += 256) {
    const int r = idx >> 6, d = idx & 63;
    q_lds[r][d] = Qf[base + (size_t)(row0 + r) * DMODEL + d];
  }

  // per-wave q A-frags for E (K = hd = 64 -> two 16x32 frags), loop-invariant
  const int wrow = row0 + w * 16;
  const v16bf qa0 = load_frag_a(Qb + base, DMODEL, wrow, 0,  lane);
  const v16bf qa1 = load_frag_a(Qb + base, DMODEL, wrow, 32, lane);

  v8f out0 = {}, out1 = {}, out2 = {}, out3 = {};
  const int mloc = (lane >> 4) << 3;
  const int nloc = lane & 15;
  const unsigned int klds0 = (unsigned int)(uintptr_t)&k_lds[0][0][0];
  const unsigned int klds1 = (unsigned int)(uintptr_t)&k_lds[1][0][0];
  const int NTILES = SEQ / 32;

  // prologue: DMA tile 0 into buffer 0
  if (w == 0) tdm_load_k_tile(Kf + base, klds0);

  for (int js = 0; js < NTILES; ++js) {
    const int j0  = js * 32;
    const int cur = js & 1;
    if (w == 0) {
      if (js + 1 < NTILES) {
        // issue next tile into the other buffer, then wait until only it
        // remains outstanding (same-wave TDM ops complete in order)
        tdm_load_k_tile(Kf + base + (size_t)(j0 + 32) * DMODEL,
                        cur ? klds0 : klds1);
        __builtin_amdgcn_s_wait_tensorcnt(1);
      } else {
        __builtin_amdgcn_s_wait_tensorcnt(0);
      }
    }
    // cooperative stage of d-major bf16 V tile (TDM cannot transpose)
    for (int idx = threadIdx.x; idx < 32 * HDIM; idx += 256) {
      const int j = idx >> 6, d = idx & 63;
      vT[d][j] = Vb[base + (size_t)(j0 + j) * DMODEL + d];
    }
    __syncthreads();

    #pragma unroll
    for (int jt = 0; jt < 2; ++jt) {
      // E = q . k^T via WMMA (two k=32 steps)
      v16bf kb0 = load_frag_b(Kb + base, DMODEL, j0 + jt * 16, 0,  lane);
      v16bf kb1 = load_frag_b(Kb + base, DMODEL, j0 + jt * 16, 32, lane);
      v8f e = {};
      e = WMMA_BF16(qa0, kb0, e);
      e = WMMA_BF16(qa1, kb1, e);

      // N = sum_d |q - k|  (f32 VALU; q reads broadcast per half-wave)
      float nacc[8] = {0.f,0.f,0.f,0.f,0.f,0.f,0.f,0.f};
      const int jrow = jt * 16 + nloc;
      for (int d = 0; d < HDIM; ++d) {
        const float kv = k_lds[cur][jrow][d];
        #pragma unroll
        for (int r = 0; r < 8; ++r)
          nacc[r] += fabsf(q_lds[w * 16 + mloc + r][d] - kv);
      }

      // coda = tanh(E) * sigmoid(N) -> bf16, per-wave LDS tile (C->A relayout)
      #pragma unroll
      for (int r = 0; r < 8; ++r) {
        const float wv = tanhf(e[r]) * (1.0f / (1.0f + __expf(-nacc[r])));
        coda_lds[w][mloc + r][jt * 16 + nloc] = to_bf16(wv);
      }
    }

    // make this wave's coda stores visible to its own frag reloads
    asm volatile("s_wait_dscnt 0" ::: "memory");

    // out(16x64) += coda(16x32) @ v(32x64)
    const v16bf ca  = load_frag_a(&coda_lds[w][0][0], 32, 0, 0, lane);
    const v16bf vb0 = load_frag_b(&vT[0][0], 40, 0,  0, lane);
    const v16bf vb1 = load_frag_b(&vT[0][0], 40, 16, 0, lane);
    const v16bf vb2 = load_frag_b(&vT[0][0], 40, 32, 0, lane);
    const v16bf vb3 = load_frag_b(&vT[0][0], 40, 48, 0, lane);
    out0 = WMMA_BF16(ca, vb0, out0);
    out1 = WMMA_BF16(ca, vb1, out1);
    out2 = WMMA_BF16(ca, vb2, out2);
    out3 = WMMA_BF16(ca, vb3, out3);
    __syncthreads();
  }

  // store O (bf16, [B,S,D] head slice) for the FC GEMM
  #pragma unroll
  for (int r = 0; r < 8; ++r) {
    const size_t ro = base + (size_t)(wrow + mloc + r) * DMODEL;
    Ob[ro +  0 + nloc] = to_bf16(out0[r]);
    Ob[ro + 16 + nloc] = to_bf16(out1[r]);
    Ob[ro + 32 + nloc] = to_bf16(out2[r]);
    Ob[ro + 48 + nloc] = to_bf16(out3[r]);
  }
}

// ---------------------------------------------------------------------------
// Per-row LayerNorm (one block per row, D = 1024)
// ---------------------------------------------------------------------------
__global__ __launch_bounds__(256) void k_layernorm(
    const float* __restrict__ X, const float* __restrict__ gamma,
    const float* __restrict__ beta, float* __restrict__ out) {
  __shared__ float red[16];
  const int row = blockIdx.x;
  const float* x = X + (size_t)row * DMODEL;

  float s = 0.f, ss = 0.f, vals[4];
  #pragma unroll
  for (int i = 0; i < 4; ++i) {
    const float v = x[threadIdx.x + i * 256];
    vals[i] = v; s += v; ss += v * v;
  }
  #pragma unroll
  for (int o = 16; o > 0; o >>= 1) { s += __shfl_xor(s, o); ss += __shfl_xor(ss, o); }
  const int w = threadIdx.x >> 5, lane = threadIdx.x & 31;
  if (lane == 0) { red[w] = s; red[8 + w] = ss; }
  __syncthreads();
  if (w == 0) {
    float a  = (lane < 8) ? red[lane]     : 0.f;
    float b2 = (lane < 8) ? red[8 + lane] : 0.f;
    #pragma unroll
    for (int o = 4; o > 0; o >>= 1) { a += __shfl_xor(a, o); b2 += __shfl_xor(b2, o); }
    if (lane == 0) { red[0] = a; red[8] = b2; }
  }
  __syncthreads();
  const float mu   = red[0] * (1.f / DMODEL);
  const float var  = red[8] * (1.f / DMODEL) - mu * mu;
  const float rstd = rsqrtf(var + 1e-5f);
  #pragma unroll
  for (int i = 0; i < 4; ++i) {
    const int d = threadIdx.x + i * 256;
    out[(size_t)row * DMODEL + d] = (vals[i] - mu) * rstd * gamma[d] + beta[d];
  }
}

// ---------------------------------------------------------------------------
extern "C" void kernel_launch(void* const* d_in, const int* in_sizes, int n_in,
                              void* d_out, int out_size, void* d_ws, size_t ws_size,
                              hipStream_t stream) {
  (void)in_sizes; (void)n_in; (void)out_size; (void)ws_size;
  const float* query = (const float*)d_in[0];
  const float* key   = (const float*)d_in[1];
  const float* value = (const float*)d_in[2];
  const float* Wq    = (const float*)d_in[3];
  const float* Wk    = (const float*)d_in[4];
  const float* Wv    = (const float*)d_in[5];
  const float* Wfc   = (const float*)d_in[6];
  const float* bfc   = (const float*)d_in[7];
  const float* gamma = (const float*)d_in[8];
  const float* beta  = (const float*)d_in[9];

  const int M  = BATCH * SEQ;        // 4096
  const int MD = M * DMODEL;         // 4,194,304
  const int DD = DMODEL * DMODEL;    // 1,048,576

  size_t off = 0;
  auto sub = [&](size_t bytes) -> void* {
    void* p = (char*)d_ws + off;
    off += (bytes + 255) & ~(size_t)255;
    return p;
  };
  bf16_t* Xq_b  = (bf16_t*)sub((size_t)MD * 2);
  bf16_t* Xk_b  = (bf16_t*)sub((size_t)MD * 2);
  bf16_t* Xv_b  = (bf16_t*)sub((size_t)MD * 2);
  bf16_t* Wq_b  = (bf16_t*)sub((size_t)DD * 2);
  bf16_t* Wk_b  = (bf16_t*)sub((size_t)DD * 2);
  bf16_t* Wv_b  = (bf16_t*)sub((size_t)DD * 2);
  bf16_t* Wfc_b = (bf16_t*)sub((size_t)DD * 2);
  float*  Qf    = (float*) sub((size_t)MD * 4);
  float*  Kf    = (float*) sub((size_t)MD * 4);
  bf16_t* Qb    = (bf16_t*)sub((size_t)MD * 2);
  bf16_t* Kb    = (bf16_t*)sub((size_t)MD * 2);
  bf16_t* Vb    = (bf16_t*)sub((size_t)MD * 2);
  bf16_t* Ob    = (bf16_t*)sub((size_t)MD * 2);
  float*  Xf    = (float*) sub((size_t)MD * 4);

  // fp32 -> bf16 conversions
  k_cvt_bf16<<<(MD + 255) / 256, 256, 0, stream>>>(query, Xq_b, MD);
  k_cvt_bf16<<<(MD + 255) / 256, 256, 0, stream>>>(key,   Xk_b, MD);
  k_cvt_bf16<<<(MD + 255) / 256, 256, 0, stream>>>(value, Xv_b, MD);
  k_cvt_bf16<<<(DD + 255) / 256, 256, 0, stream>>>(Wq,  Wq_b,  DD);
  k_cvt_bf16<<<(DD + 255) / 256, 256, 0, stream>>>(Wk,  Wk_b,  DD);
  k_cvt_bf16<<<(DD + 255) / 256, 256, 0, stream>>>(Wv,  Wv_b,  DD);
  k_cvt_bf16<<<(DD + 255) / 256, 256, 0, stream>>>(Wfc, Wfc_b, DD);

  // q/k/v projections (x @ W^T); block tile 64x128
  dim3 gg(DMODEL / 128, M / 64);
  k_gemm_nt<<<gg, 256, 0, stream>>>(Xq_b, Wq_b, M, DMODEL, DMODEL,
                                    Qf, Qb, nullptr, nullptr);
  k_gemm_nt<<<gg, 256, 0, stream>>>(Xk_b, Wk_b, M, DMODEL, DMODEL,
                                    Kf, Kb, nullptr, nullptr);
  k_gemm_nt<<<gg, 256, 0, stream>>>(Xv_b, Wv_b, M, DMODEL, DMODEL,
                                    nullptr, Vb, nullptr, nullptr);

  // fused CoDA attention
  dim3 ga(SEQ / 128, BATCH * NHEADS);
  k_coda_attn<<<ga, 256, 0, stream>>>(Qf, Kf, Qb, Kb, Vb, Ob);

  // fc + bias + residual
  k_gemm_nt<<<gg, 256, 0, stream>>>(Ob, Wfc_b, M, DMODEL, DMODEL,
                                    Xf, nullptr, bfc, query);

  // layernorm -> d_out
  k_layernorm<<<M, 256, 0, stream>>>(Xf, gamma, beta, (float*)d_out);
}
